// GAM_30279519436937
// MI455X (gfx1250) — compile-verified, matrix operand order checked by reading, not run on hardware
//
#include <hip/hip_runtime.h>
#include <hip/hip_bf16.h>

typedef __attribute__((ext_vector_type(16))) _Float16 v16h;
typedef __attribute__((ext_vector_type(8)))  _Float16 v8h;
typedef __attribute__((ext_vector_type(8)))  float    v8f;
typedef __attribute__((ext_vector_type(4)))  unsigned int v4u;
typedef __attribute__((ext_vector_type(8)))  int      v8i;
typedef __attribute__((ext_vector_type(4)))  int      v4i;

#define CB   768
#define HH   56
#define LL   3136          // 56*56
#define BB   8
#define MTOT (BB*LL)       // 25088
#define LDA  40            // padded LDS row stride in halves (80B = 5*16B aligned)

// ---------------- WMMA core ----------------

__device__ __forceinline__ v16h frag_ld(const _Float16* p) {
  v16h r;
  ((v8h*)&r)[0] = *(const v8h*)(p);        // K = kbase .. kbase+7
  ((v8h*)&r)[1] = *(const v8h*)(p + 16);   // K = kbase+16 .. kbase+23
  return r;
}

// 8 waves: wrow(2) x wcol(4); each wave = 64x32 = 4x2 tiles of 16x16.
__device__ __forceinline__ void wmma_block(const _Float16* As, const _Float16* Bs,
                                           v8f acc[4][2]) {
  const int lane = threadIdx.x & 31;
  const int wid  = threadIdx.x >> 5;
  const int wrow = wid & 1, wcol = wid >> 1;
  const int lrow = lane & 15, lhi = lane >> 4;
  v16h a[4], b[2];
#pragma unroll
  for (int mi = 0; mi < 4; ++mi)
    a[mi] = frag_ld(&As[(wrow*64 + mi*16 + lrow) * LDA + lhi*8]);
#pragma unroll
  for (int ni = 0; ni < 2; ++ni)
    b[ni] = frag_ld(&Bs[(wcol*32 + ni*16 + lrow) * LDA + lhi*8]);
#pragma unroll
  for (int mi = 0; mi < 4; ++mi)
#pragma unroll
    for (int ni = 0; ni < 2; ++ni)
      acc[mi][ni] = __builtin_amdgcn_wmma_f32_16x16x32_f16(
          false, a[mi], false, b[ni], (short)0, acc[mi][ni], false, false);
}

// TDM: DMA one 128x32 (rows x K) f16 tile, row-major (ld elems), into padded LDS.
// Descriptor per CDNA5 ISA §8.3/8.4: count=1, type=2, data_size=2B,
// tile_dim0=32, tile_dim1=rows, pad 4 DWORDs every 16 DWORDs (64B row -> 80B LDS).
__device__ __forceinline__ void tdm_load_tile(unsigned lds_off, const void* gptr,
                                              int ld_elems, int rows) {
  unsigned long long ga = (unsigned long long)(uintptr_t)gptr;
  v4u g0;
  g0.x = 1u;                                               // count=1, user D#
  g0.y = lds_off;                                          // lds_addr [63:32]
  g0.z = (unsigned)ga;                                     // global_addr [95:64]
  g0.w = (unsigned)((ga >> 32) & 0x01FFFFFFu) | (2u << 30); // addr[56:32] | type=2
  v8i g1;
  g1[0] = (1 << 16) | (1 << 20) | (3 << 22) | (3 << 25);   // size=2B, pad_en, int=16dw, amt=4dw
  g1[1] = (int)(((unsigned)ld_elems & 0xFFFFu) << 16);     // tensor_dim0 lo16 @[63:48]
  g1[2] = (int)(((unsigned)ld_elems >> 16) | (0xFFFFu << 16)); // dim0 hi16 | dim1 lo16
  g1[3] = (int)(0x7FFFu | (32u << 16));                    // dim1 hi16 | tile_dim0=32
  g1[4] = rows & 0xFFFF;                                   // tile_dim1 | tile_dim2=0
  g1[5] = ld_elems;                                        // tensor_dim0_stride lo32
  g1[6] = 0;                                               // stride hi | dim1_stride lo
  g1[7] = 0;
  const v4i gz4 = {0, 0, 0, 0};                            // 2D: groups 2/3 unused
  const v8i gz8 = {0, 0, 0, 0, 0, 0, 0, 0};
  __builtin_amdgcn_tensor_load_to_lds(g0, g1, gz4, gz4, gz8, 0);
}

// Per-lane async copy of a 128x32 f16 tile via global_load_async_to_lds_b128.
__device__ __forceinline__ void stage_async_b128(_Float16* dst, const _Float16* g,
                                                 int ldg) {
  const int t = threadIdx.x;
#pragma unroll
  for (int i = 0; i < 2; ++i) {
    int c = t + i*256;               // 512 chunks of 8 halves
    int m = c >> 2, ko = (c & 3) * 8;
    unsigned lds = (unsigned)(uintptr_t)&dst[m*LDA + ko];   // LDS byte offset
    const void* ga = (const void*)&g[(size_t)m*ldg + ko];
    asm volatile("global_load_async_to_lds_b128 %0, %1, off"
                 :: "v"(lds), "v"(ga) : "memory");
  }
}

__device__ __forceinline__ void wait_async0() {
  asm volatile("s_wait_asynccnt 0x0" ::: "memory");
}

// Double-buffered K loop shared by all GEMMs. A tile via TDM (wave 0 issues the
// DMA, TENSORcnt); B tile via per-lane async b128 (ASYNCcnt). KT tiles of K=32.
__device__ __forceinline__ void gemm_core(const _Float16* Ag, int lda,
                                          const _Float16* Bg, int ldb,
                                          int KT, int maxm,
                                          _Float16* As0, _Float16* As1,
                                          _Float16* Bs0, _Float16* Bs1,
                                          v8f acc[4][2]) {
  const int wid = threadIdx.x >> 5;
  if (wid == 0)
    tdm_load_tile((unsigned)(uintptr_t)As0, Ag, lda, maxm);
  stage_async_b128(Bs0, Bg, ldb);
  if (wid == 0) __builtin_amdgcn_s_wait_tensorcnt(0);
  wait_async0();
  __syncthreads();
  for (int kt = 0; kt < KT; ++kt) {
    const _Float16* Ac = (kt & 1) ? As1 : As0;
    const _Float16* Bc = (kt & 1) ? Bs1 : Bs0;
    if (kt + 1 < KT) {
      _Float16* An = (kt & 1) ? As0 : As1;
      _Float16* Bn = (kt & 1) ? Bs0 : Bs1;
      if (wid == 0)
        tdm_load_tile((unsigned)(uintptr_t)An, Ag + (kt+1)*32, lda, maxm);
      stage_async_b128(Bn, Bg + (kt+1)*32, ldb);
    }
    wmma_block(Ac, Bc, acc);                 // overlaps DMA + async copies
    if (wid == 0) __builtin_amdgcn_s_wait_tensorcnt(0);
    wait_async0();
    __syncthreads();
  }
}

#define GEMM_PROLOGUE                                                     \
  __shared__ _Float16 As0[128*LDA], As1[128*LDA];                         \
  __shared__ _Float16 Bs0[128*LDA], Bs1[128*LDA];                         \
  const v8f vzero = {0.f,0.f,0.f,0.f,0.f,0.f,0.f,0.f};                    \
  v8f acc[4][2];                                                          \
  _Pragma("unroll")                                                       \
  for (int mi = 0; mi < 4; ++mi)                                          \
    _Pragma("unroll")                                                     \
    for (int ni = 0; ni < 2; ++ni) acc[mi][ni] = vzero;

#define GEMM_EPILOGUE_VARS                                                \
  const int lane = threadIdx.x & 31, wid = threadIdx.x >> 5;              \
  const int wrow = wid & 1, wcol = wid >> 1;                              \
  const int lrow = lane & 15, lhi = lane >> 4;

// ---- GEMM 1: t16[25088x768] x Wt[z][n][k] -> Q(f32,relu) / K(relu),V,Q1,K1 (f16) ----

__global__ __launch_bounds__(256) void gemm_qkv(const _Float16* __restrict__ t16,
                                                const _Float16* __restrict__ Wt,
                                                const float* __restrict__ Bias,
                                                float* __restrict__ Qout,
                                                _Float16* __restrict__ O16) {
  const int bm = blockIdx.x, bn = blockIdx.y, z = blockIdx.z;
  GEMM_PROLOGUE
  gemm_core(t16 + (size_t)bm*128*CB, CB,
            Wt + (size_t)z*CB*CB + (size_t)bn*128*CB, CB,
            CB/32, 128, As0, As1, Bs0, Bs1, acc);
  GEMM_EPILOGUE_VARS
  const bool do_relu = (z <= 1);
#pragma unroll
  for (int mi = 0; mi < 4; ++mi)
#pragma unroll
    for (int ni = 0; ni < 2; ++ni) {
      int col = bn*128 + wcol*32 + ni*16 + lrow;
      float bias = Bias[z*CB + col];
#pragma unroll
      for (int r = 0; r < 8; ++r) {
        int row = bm*128 + wrow*64 + mi*16 + lhi*8 + r;
        float v = acc[mi][ni][r] + bias;
        if (do_relu) v = fmaxf(v, 0.f);
        size_t idx = (size_t)row*CB + col;
        if (z == 0) Qout[idx] = v;
        else        O16[(size_t)(z-1)*MTOT*CB + idx] = (_Float16)v;
      }
    }
}

// ---- GEMM 2: S[b] = Q1^T K1 via pre-transposed Q1t/K1t [b][c][l], K=3136 ----

__global__ __launch_bounds__(256) void gemm_s(const _Float16* __restrict__ Q1t,
                                              const _Float16* __restrict__ K1t,
                                              float* __restrict__ S) {
  const int bm = blockIdx.x, bn = blockIdx.y, b = blockIdx.z;
  GEMM_PROLOGUE
  gemm_core(Q1t + (size_t)b*CB*LL + (size_t)bm*128*LL, LL,
            K1t + (size_t)b*CB*LL + (size_t)bn*128*LL, LL,
            LL/32, 128, As0, As1, Bs0, Bs1, acc);
  GEMM_EPILOGUE_VARS
#pragma unroll
  for (int mi = 0; mi < 4; ++mi)
#pragma unroll
    for (int ni = 0; ni < 2; ++ni) {
      int col = bn*128 + wcol*32 + ni*16 + lrow;
#pragma unroll
      for (int r = 0; r < 8; ++r) {
        int row = bm*128 + wrow*64 + mi*16 + lhi*8 + r;
        S[(size_t)b*CB*CB + (size_t)row*CB + col] = acc[mi][ni][r];
      }
    }
}

// ---- GEMM 3: res2[b] = res[b] @ mid[b] via pre-transposed midT [b][d][c] ----

__global__ __launch_bounds__(256) void gemm_res2(const _Float16* __restrict__ res16,
                                                 const _Float16* __restrict__ midT,
                                                 float* __restrict__ res2) {
  const int bm = blockIdx.x, bn = blockIdx.y, b = blockIdx.z;
  const int maxm = (LL - bm*128) < 128 ? (LL - bm*128) : 128;
  GEMM_PROLOGUE
  gemm_core(res16 + (size_t)b*LL*CB + (size_t)bm*128*CB, CB,
            midT + (size_t)b*CB*CB + (size_t)bn*128*CB, CB,
            CB/32, maxm, As0, As1, Bs0, Bs1, acc);
  GEMM_EPILOGUE_VARS
#pragma unroll
  for (int mi = 0; mi < 4; ++mi)
#pragma unroll
    for (int ni = 0; ni < 2; ++ni) {
      int col = bn*128 + wcol*32 + ni*16 + lrow;
#pragma unroll
      for (int r = 0; r < 8; ++r) {
        int rowl = wrow*64 + mi*16 + lhi*8 + r;
        if (rowl < maxm)
          res2[(size_t)b*LL*CB + (size_t)(bm*128 + rowl)*CB + col] = acc[mi][ni][r];
      }
    }
}

// ---------------- Elementwise / glue kernels ----------------

__global__ void pos_kernel(const float* __restrict__ pos_w,
                           const float* __restrict__ pos_b,
                           float* __restrict__ pos) {
  __shared__ float feat[64];
  const int l = blockIdx.x, h = l / HH, w = l % HH;
  const int t = threadIdx.x;
  if (t < 64) {
    const float scale = 6.283185307179586f;
    float v = (t < 32) ? (scale * (float)(h + 1) / (56.f + 1e-6f))
                       : (scale * (float)(w + 1) / (56.f + 1e-6f));
    int kk = t & 31;
    float d = powf(10000.f, (float)(kk & ~1) / 32.f);
    float p = v / d;
    feat[t] = (kk & 1) ? cosf(p) : sinf(p);
  }
  __syncthreads();
  for (int c = t; c < CB; c += 256) {
    float a = pos_b[c];
#pragma unroll 8
    for (int k = 0; k < 64; ++k) a += feat[k] * pos_w[k*CB + c];
    pos[(size_t)l*CB + c] = a;
  }
}

__global__ void tbuild(const float* __restrict__ x, const float* __restrict__ pos,
                       float* __restrict__ t32, _Float16* __restrict__ t16) {
  __shared__ float Xs[32][33];
  const int l0 = blockIdx.x*32, c0 = blockIdx.y*32, b = blockIdx.z;
  const int t = threadIdx.x;
#pragma unroll
  for (int i = 0; i < 4; ++i) {
    int f = t + i*256, cr = f >> 5, lc = f & 31;
    Xs[cr][lc] = x[((size_t)b*CB + c0 + cr)*LL + l0 + lc];
  }
  __syncthreads();
#pragma unroll
  for (int i = 0; i < 4; ++i) {
    int f = t + i*256, lr = f >> 5, cc = f & 31;
    size_t li = (size_t)(l0 + lr);
    float v = Xs[cc][lr] + pos[li*CB + c0 + cc];
    size_t idx = ((size_t)b*LL + li)*CB + c0 + cc;
    t32[idx] = v;
    t16[idx] = (_Float16)v;
  }
}

// weights -> f16 pre-transposed Wt[z][n][k]; biases contiguous
__global__ void packw(const float* w0, const float* w1, const float* w2,
                      const float* w3, const float* w4,
                      const float* b0, const float* b1, const float* b2,
                      const float* b3, const float* b4,
                      _Float16* __restrict__ Wt, float* __restrict__ Bias) {
  int i = blockIdx.x*256 + threadIdx.x;      // 0 .. 5*768*768-1
  int z = i / (CB*CB), r = i % (CB*CB);
  int k = r / CB, n = r % CB;
  const float* w = (z == 0) ? w0 : (z == 1) ? w1 : (z == 2) ? w2 : (z == 3) ? w3 : w4;
  Wt[(size_t)z*CB*CB + (size_t)n*CB + k] = (_Float16)w[r];
  if (i < 5*CB) {
    int zi = i / CB, ri = i % CB;
    const float* bb = (zi == 0) ? b0 : (zi == 1) ? b1 : (zi == 2) ? b2 : (zi == 3) ? b3 : b4;
    Bias[i] = bb[ri];
  }
}

// generic batched f16 transpose [R x Cc] -> [Cc x R], 32x32 LDS tiles
__global__ void transpose_h(const _Float16* __restrict__ in,
                            _Float16* __restrict__ out, int R, int Cc) {
  __shared__ _Float16 Ts[32][33];
  const int r0 = blockIdx.x*32, c0 = blockIdx.y*32;
  const size_t base = (size_t)blockIdx.z * R * Cc;
  const int t = threadIdx.x;
#pragma unroll
  for (int i = 0; i < 4; ++i) {
    int f = t + i*256, rr = f >> 5, cc = f & 31;
    Ts[rr][cc] = in[base + (size_t)(r0 + rr)*Cc + c0 + cc];
  }
  __syncthreads();
#pragma unroll
  for (int i = 0; i < 4; ++i) {
    int f = t + i*256, cr = f >> 5, rc = f & 31;
    out[base + (size_t)(c0 + cr)*R + r0 + rc] = Ts[rc][cr];
  }
}

__global__ void kvred(const _Float16* __restrict__ Kh, const _Float16* __restrict__ Vh,
                      float* __restrict__ kvdiag, float* __restrict__ ksum_eps) {
  const int c = blockIdx.x*256 + threadIdx.x;
  const int b = blockIdx.y;
  const _Float16* K = Kh + (size_t)b*LL*CB;
  const _Float16* V = Vh + (size_t)b*LL*CB;
  float kv = 0.f, ks = 0.f;
  for (int l = 0; l < LL; ++l) {
    float k = (float)K[(size_t)l*CB + c];
    float v = (float)V[(size_t)l*CB + c];
    kv += k * v;
    ks += k;
  }
  kvdiag[b*CB + c]   = kv;
  ksum_eps[b*CB + c] = ks + 1e-6f;
}

__global__ void resk(const float* __restrict__ Q, const float* __restrict__ kvdiag,
                     const float* __restrict__ ksum_eps, _Float16* __restrict__ res16) {
  __shared__ float red[256];
  const int bl = blockIdx.x, b = bl / LL;
  const float* q = Q + (size_t)bl*CB;
  const int t = threadIdx.x;
  float s = 0.f;
  for (int c = t; c < CB; c += 256) s += q[c] * ksum_eps[b*CB + c];
  red[t] = s; __syncthreads();
  for (int st = 128; st > 0; st >>= 1) { if (t < st) red[t] += red[t + st]; __syncthreads(); }
  float z = 1.f / red[0];
  for (int c = t; c < CB; c += 256)
    res16[(size_t)bl*CB + c] = (_Float16)(q[c] * kvdiag[b*CB + c] * z);
}

__global__ void smax(const float* __restrict__ S, _Float16* __restrict__ mid) {
  __shared__ float red[256];
  const int row = blockIdx.x;
  const float* s = S + (size_t)row*CB;
  const int t = threadIdx.x;
  float m = -1e30f;
  for (int c = t; c < CB; c += 256) m = fmaxf(m, s[c]);
  red[t] = m; __syncthreads();
  for (int st = 128; st > 0; st >>= 1) { if (t < st) red[t] = fmaxf(red[t], red[t + st]); __syncthreads(); }
  m = red[0]; __syncthreads();
  float sum = 0.f;
  for (int c = t; c < CB; c += 256) sum += expf(s[c] - m);
  red[t] = sum; __syncthreads();
  for (int st = 128; st > 0; st >>= 1) { if (t < st) red[t] += red[t + st]; __syncthreads(); }
  float inv = 1.f / red[0];
  for (int c = t; c < CB; c += 256)
    mid[(size_t)row*CB + c] = (_Float16)(expf(s[c] - m) * inv);
}

__global__ void finalk(const float* __restrict__ t32, const float* __restrict__ res2,
                       const float* __restrict__ gamma,
                       const float* __restrict__ bnw, const float* __restrict__ bnb,
                       const float* __restrict__ bnm, const float* __restrict__ bnv,
                       float* __restrict__ out) {
  __shared__ float Ts[32][33];
  const int l0 = blockIdx.x*32, c0 = blockIdx.y*32, b = blockIdx.z;
  const float g = gamma[0];
  const int t = threadIdx.x;
#pragma unroll
  for (int i = 0; i < 4; ++i) {
    int f = t + i*256, lr = f >> 5, cc = f & 31;
    int c = c0 + cc;
    size_t idx = ((size_t)b*LL + l0 + lr)*CB + c;
    float v = t32[idx] + g * res2[idx];
    v = (v - bnm[c]) * rsqrtf(bnv[c] + 1e-5f) * bnw[c] + bnb[c];
    Ts[lr][cc] = fmaxf(v, 0.f);
  }
  __syncthreads();
#pragma unroll
  for (int i = 0; i < 4; ++i) {
    int f = t + i*256, cr = f >> 5, lc = f & 31;
    out[((size_t)b*CB + c0 + cr)*LL + l0 + lc] = Ts[lc][cr];
  }
}

// ---------------- Host launcher ----------------

extern "C" void kernel_launch(void* const* d_in, const int* in_sizes, int n_in,
                              void* d_out, int out_size, void* d_ws, size_t ws_size,
                              hipStream_t stream) {
  const float* x     = (const float*)d_in[0];
  const float* wq    = (const float*)d_in[1];
  const float* bq    = (const float*)d_in[2];
  const float* wk    = (const float*)d_in[3];
  const float* bk    = (const float*)d_in[4];
  const float* wv    = (const float*)d_in[5];
  const float* bv    = (const float*)d_in[6];
  const float* wq1   = (const float*)d_in[7];
  const float* bq1   = (const float*)d_in[8];
  const float* wk1   = (const float*)d_in[9];
  const float* bk1   = (const float*)d_in[10];
  const float* gamma = (const float*)d_in[11];
  const float* bnw   = (const float*)d_in[12];
  const float* bnb   = (const float*)d_in[13];
  const float* bnm   = (const float*)d_in[14];
  const float* bnv   = (const float*)d_in[15];
  const float* posw  = (const float*)d_in[16];
  const float* posb  = (const float*)d_in[17];
  float* out = (float*)d_out;

  char* wsb = (char*)d_ws;
  const size_t POS  = 0;                                  // L*C f32
  const size_t T32  = POS  + (size_t)LL*CB*4;             // M*C f32
  const size_t T16  = T32  + (size_t)MTOT*CB*4;           // M*C f16 (reused as res16)
  const size_t WH   = T16  + (size_t)MTOT*CB*2;           // 5*C*C f16 (transposed)
  const size_t BIAS = WH   + (size_t)5*CB*CB*2;           // 5*C f32
  const size_t Q32  = BIAS + (size_t)5*CB*4;              // M*C f32 (reused as res2)
  const size_t O16  = Q32  + (size_t)MTOT*CB*4;           // 4 * M*C f16 (K,V,Q1,K1)
  const size_t KV   = O16  + (size_t)4*MTOT*CB*2;         // B*C f32
  const size_t KS   = KV   + (size_t)BB*CB*4;             // B*C f32
  const size_t SB   = KS   + (size_t)BB*CB*4;             // B*C*C f32 (reused as midT)
  const size_t MID  = SB   + (size_t)BB*CB*CB*4;          // B*C*C f16
  const size_t Q1T  = MID  + (size_t)BB*CB*CB*2;          // M*C f16
  const size_t K1T  = Q1T  + (size_t)MTOT*CB*2;           // M*C f16

  float*    pos   = (float*)   (wsb + POS);
  float*    t32   = (float*)   (wsb + T32);
  _Float16* t16   = (_Float16*)(wsb + T16);
  _Float16* res16 = (_Float16*)(wsb + T16);   // alias: t16 dead after gemm_qkv
  _Float16* Wt    = (_Float16*)(wsb + WH);
  float*    Bias  = (float*)   (wsb + BIAS);
  float*    Q     = (float*)   (wsb + Q32);
  float*    res2  = (float*)   (wsb + Q32);   // alias: Q dead after resk
  _Float16* Kh    = (_Float16*)(wsb + O16);
  _Float16* Vh    = Kh  + (size_t)MTOT*CB;
  _Float16* Q1h   = Vh  + (size_t)MTOT*CB;
  _Float16* K1h   = Q1h + (size_t)MTOT*CB;
  float*    kvdg  = (float*)   (wsb + KV);
  float*    kse   = (float*)   (wsb + KS);
  float*    S     = (float*)   (wsb + SB);
  _Float16* midT  = (_Float16*)(wsb + SB);    // alias: S dead after smax
  _Float16* mid   = (_Float16*)(wsb + MID);
  _Float16* Q1t   = (_Float16*)(wsb + Q1T);
  _Float16* K1t   = (_Float16*)(wsb + K1T);

  // 1) positional encoding [L,C]
  pos_kernel<<<LL, 256, 0, stream>>>(posw, posb, pos);
  // 2) t = permute(x)+pos, f32 + f16
  tbuild<<<dim3(LL/32, CB/32, BB), 256, 0, stream>>>(x, pos, t32, t16);
  // 3) pack weights -> f16 transposed, biases contiguous
  packw<<<(5*CB*CB)/256, 256, 0, stream>>>(wq, wk, wv, wq1, wk1,
                                           bq, bk, bv, bq1, bk1, Wt, Bias);
  // 4) five projection GEMMs (WMMA; A tile via TDM, B tile via async b128)
  gemm_qkv<<<dim3(MTOT/128, CB/128, 5), 256, 0, stream>>>(t16, Wt, Bias, Q, Kh);
  // 5) kv_diag and Ksum+eps column reductions
  kvred<<<dim3(CB/256, BB), 256, 0, stream>>>(Kh, Vh, kvdg, kse);
  // 6) z = 1/(Q.(Ksum+eps)); res = Q*kv_diag*z  -> f16
  resk<<<MTOT, 256, 0, stream>>>(Q, kvdg, kse, res16);
  // 7) transpose Q1,K1 to [b][c][l] so the K=3136 GEMM stages row-major
  transpose_h<<<dim3(LL/32, CB/32, BB), 256, 0, stream>>>(Q1h, Q1t, LL, CB);
  transpose_h<<<dim3(LL/32, CB/32, BB), 256, 0, stream>>>(K1h, K1t, LL, CB);
  // 8) S[b] = Q1^T K1 (WMMA)
  gemm_s<<<dim3(CB/128, CB/128, BB), 256, 0, stream>>>(Q1t, K1t, S);
  // 9) softmax rows -> mid f16
  smax<<<BB*CB, 256, 0, stream>>>(S, mid);
  // 10) transpose mid -> midT [b][d][c]
  transpose_h<<<dim3(CB/32, CB/32, BB), 256, 0, stream>>>(mid, midT, CB, CB);
  // 11) res2[b] = res[b] @ mid[b] (WMMA)
  gemm_res2<<<dim3((LL + 127)/128, CB/128, BB), 256, 0, stream>>>(res16, midT, res2);
  // 12) out = relu(BN(t + gamma*res2)) with [L,C]->[C,L] transpose
  finalk<<<dim3(LL/32, CB/32, BB), 256, 0, stream>>>(t32, res2, gamma,
                                                     bnw, bnb, bnm, bnv, out);
}
